// SelfAttention_19387482374577
// MI455X (gfx1250) — matrix operand order repaired and checked by reading, not compiled, not run on hardware
//
#include <hip/hip_runtime.h>
#include <hip/hip_bf16.h>

// ---------------------------------------------------------------------------
// Causal multi-head self-attention for MI455X (gfx1250), all matmuls on WMMA.
// B=2, S=2048, H=16, d_head=64, d_model=1024.
// Pattern (536 MB) + attn_out_by_head (268 MB) streamed with TH=NT stores;
// out-projection and head-sum fused to avoid a 268 MB HBM round-trip.
// ---------------------------------------------------------------------------

typedef __attribute__((ext_vector_type(16))) __bf16 bf16x16;
typedef __attribute__((ext_vector_type(8)))  __bf16 bf16x8;
typedef __attribute__((ext_vector_type(8)))  float  f32x8;
typedef __attribute__((ext_vector_type(4)))  float  f32x4;

#define DMODEL 1024
#define DHEAD  64
#define NHEADS 16
#define SEQ    2048
#define BATCH  2
#define MROWS  (BATCH * SEQ)           // 4096
#define LDP    2052                    // padded LDS row pitch (floats), kills bank conflicts
#define MASKF  (-100000.0f)

// ---- WMMA helpers ----------------------------------------------------------

__device__ __forceinline__ f32x8 wmma_bf16(bf16x16 a, bf16x16 b, f32x8 c) {
  // D = A(16x32 bf16) * B(32x16 bf16) + C(16x16 f32)
  return __builtin_amdgcn_wmma_f32_16x16x32_bf16(false, a, false, b, (short)0, c,
                                                 false, false);
}

// A fragment 16x32 (bf16, row-major source, ld in elements).
// ISA 7.12.2: lane m = lane&15; lanes 0-15 hold K=0..7 & 16..23, lanes 16-31 K=8..15 & 24..31.
__device__ __forceinline__ bf16x16 load_a_global(const __bf16* __restrict__ base,
                                                 int ld, int k0, int lane) {
  const int m  = lane & 15;
  const int kb = (lane >> 4) << 3;     // 0 or 8
  const __bf16* p = base + (size_t)m * ld + k0 + kb;
  bf16x8 lo = *(const bf16x8*)(p);          // K = k0+kb .. +7
  bf16x8 hi = *(const bf16x8*)(p + 16);     // K = k0+16+kb .. +7
  bf16x16 f;
#pragma unroll
  for (int i = 0; i < 8; ++i) { f[i] = lo[i]; f[i + 8] = hi[i]; }
  return f;
}

// B fragment 32x16 (bf16), source stored K-contiguous per column (ld = column stride).
// lanes 0-15: col n, K=0..7 & 16..23 ; lanes 16-31: col n, K=8..15 & 24..31.
__device__ __forceinline__ bf16x16 load_b_global(const __bf16* __restrict__ base,
                                                 int ld, int k0, int lane) {
  const int n  = lane & 15;
  const int kb = (lane >> 4) << 3;     // 0 or 8
  const __bf16* p = base + (size_t)n * ld + k0 + kb;
  bf16x8 lo = *(const bf16x8*)(p);
  bf16x8 hi = *(const bf16x8*)(p + 16);
  bf16x16 f;
#pragma unroll
  for (int i = 0; i < 8; ++i) { f[i] = lo[i]; f[i + 8] = hi[i]; }
  return f;
}

// A fragment sourced from f32 LDS (softmax pattern), converted to bf16 inline.
__device__ __forceinline__ bf16x16 load_a_lds_f32(const float* __restrict__ lds,
                                                  int ld, int k0, int lane) {
  const int m  = lane & 15;
  const int kb = (lane >> 4) << 3;
  const float* p = lds + m * ld + k0 + kb;
  bf16x16 f;
#pragma unroll
  for (int i = 0; i < 8; ++i) { f[i] = (__bf16)p[i]; f[i + 8] = (__bf16)p[16 + i]; }
  return f;
}

// ---- Stage 0: precision conversion / weight transposes ---------------------

__global__ void k_cvt_x(const float* __restrict__ x, __bf16* __restrict__ xb, int n) {
  int i = blockIdx.x * blockDim.x + threadIdx.x;
  if (i < n) xb[i] = (__bf16)x[i];
}

// W_{Q,K,V}: [h, d(1024), e(64)]  ->  bf16 [h, e, d]  (K=d contiguous for B-frags)
__global__ void k_tr_wqkv(const float* __restrict__ w, __bf16* __restrict__ wt) {
  int i = blockIdx.x * blockDim.x + threadIdx.x;   // 16*1024*64
  int h = i >> 16, r = i & 65535, d = r >> 6, e = r & 63;
  wt[(size_t)h * 65536 + (size_t)e * 1024 + d] = (__bf16)w[i];
}

// W_O: [h, e(64), d(1024)]  ->  bf16 [h, d, e]  (K=e contiguous for B-frags)
__global__ void k_tr_wo(const float* __restrict__ w, __bf16* __restrict__ wt) {
  int i = blockIdx.x * blockDim.x + threadIdx.x;   // 16*64*1024
  int h = i >> 16, r = i & 65535, e = r >> 10, d = r & 1023;
  wt[(size_t)h * 65536 + (size_t)d * 64 + e] = (__bf16)w[i];
}

// ---- Stage 1: QKV projection GEMM ------------------------------------------
// C[m, h, e] = sum_d x[m,d] * W[h,d,e] + bias[h,e]
// block = 128 threads (4 waves), each wave owns one 16-wide e-tile; 16-row m tile.
// transposed==0 -> out [b,h,s,64]; transposed==1 -> out [b,h,64,S] (for V).
__global__ void k_qkv(const __bf16* __restrict__ xb, const __bf16* __restrict__ wT,
                      const float* __restrict__ bias, __bf16* __restrict__ out,
                      int transposed) {
  const int lane = threadIdx.x & 31;
  const int wave = threadIdx.x >> 5;        // 0..3
  const int m0 = blockIdx.x * 16;           // never crosses batch: S%16==0
  const int h  = blockIdx.y;
  const int b  = m0 >> 11;
  const int s0 = m0 & (SEQ - 1);
  const int e0 = wave * 16;

  const __bf16* aBase = xb + (size_t)m0 * DMODEL;
  const __bf16* bBase = wT + (size_t)h * (DMODEL * DHEAD) + (size_t)e0 * DMODEL;

  f32x8 acc = {};
#pragma unroll 4
  for (int k0 = 0; k0 < DMODEL; k0 += 32) {
    bf16x16 a  = load_a_global(aBase, DMODEL, k0, lane);
    bf16x16 bb = load_b_global(bBase, DMODEL, k0, lane);
    acc = wmma_bf16(a, bb, acc);
  }

  const int n  = lane & 15;
  const int e  = e0 + n;
  const int mh = (lane >> 4) << 3;
  const float bia = bias[h * DHEAD + e];
  if (!transposed) {
    __bf16* o = out + ((size_t)(b * NHEADS + h) * SEQ + s0) * DHEAD + e;
#pragma unroll
    for (int r = 0; r < 8; ++r)
      o[(size_t)(mh + r) * DHEAD] = (__bf16)(acc[r] + bia);
  } else {
    __bf16* o = out + ((size_t)(b * NHEADS + h) * DHEAD + e) * SEQ + s0;
#pragma unroll
    for (int r = 0; r < 8; ++r)
      o[mh + r] = (__bf16)(acc[r] + bia);
  }
}

// ---- Stage 2: scores + causal softmax + P*V --------------------------------
// One block per (b, h, 16-query tile). 128 threads = 4 waves.
// Full 16x2048 f32 score strip lives in LDS (131 KB -> 2 blocks/WGP of 320 KB).
__global__ void k_attn(const __bf16* __restrict__ qg, const __bf16* __restrict__ kg,
                       const __bf16* __restrict__ vg, float* __restrict__ pattern,
                       __bf16* __restrict__ zb) {
  extern __shared__ float sc[];             // [16][LDP] + rowinv[16]
  float* rowinv = sc + 16 * LDP;
  const int lane = threadIdx.x & 31;
  const int wave = threadIdx.x >> 5;        // 0..3
  const int q0 = blockIdx.x * 16;
  const int h  = blockIdx.y;
  const int b  = blockIdx.z;
  const size_t bh = (size_t)(b * NHEADS + h);

  // Q tile A-fragments (reused for all 128 key tiles)
  const __bf16* qBase = qg + (bh * SEQ + q0) * DHEAD;
  bf16x16 aq0 = load_a_global(qBase, DHEAD, 0,  lane);
  bf16x16 aq1 = load_a_global(qBase, DHEAD, 32, lane);

  // --- scores = Q K^T / sqrt(d) with causal mask, into LDS -----------------
  const float scale = 0.125f;               // 1/sqrt(64)
  const int n  = lane & 15;
  const int mh = (lane >> 4) << 3;
  for (int kt = wave; kt < SEQ / 16; kt += 4) {
    const int c0 = kt * 16;
    const __bf16* bBase = kg + (bh * SEQ + c0) * DHEAD;   // col = key, K = e
    bf16x16 b0 = load_b_global(bBase, DHEAD, 0,  lane);
    bf16x16 b1 = load_b_global(bBase, DHEAD, 32, lane);
    f32x8 acc = {};
    acc = wmma_bf16(aq0, b0, acc);
    acc = wmma_bf16(aq1, b1, acc);
    const int key = c0 + n;
#pragma unroll
    for (int r = 0; r < 8; ++r) {
      const int q = q0 + mh + r;
      float s = acc[r] * scale;
      if (key > q) s = MASKF;               // causal mask (matches reference MASK_FILL)
      sc[(mh + r) * LDP + key] = s;
    }
  }
  __syncthreads();

  // --- softmax reductions: 16 rows x 8 lanes each --------------------------
  {
    const int row = threadIdx.x >> 3;
    const int sub = threadIdx.x & 7;
    float* rp = sc + row * LDP;
    float mx = -3.0e38f;
    for (int c = sub; c < SEQ; c += 8) mx = fmaxf(mx, rp[c]);
#pragma unroll
    for (int o = 4; o > 0; o >>= 1) mx = fmaxf(mx, __shfl_xor(mx, o, 8));
    float sum = 0.f;
    for (int c = sub; c < SEQ; c += 8) { float e = __expf(rp[c] - mx); rp[c] = e; sum += e; }
#pragma unroll
    for (int o = 4; o > 0; o >>= 1) sum += __shfl_xor(sum, o, 8);
    if (sub == 0) rowinv[row] = 1.0f / sum;
  }
  __syncthreads();

  // --- normalize in LDS + coalesced NT pattern write (wave owns 4 rows) ----
#pragma unroll
  for (int r4 = 0; r4 < 4; ++r4) {
    const int row = wave * 4 + r4;
    const float inv = rowinv[row];
    float* rp = sc + row * LDP;
    float* gp = pattern + (bh * SEQ + q0 + row) * SEQ;
    for (int c = lane * 4; c < SEQ; c += 128) {   // 512B per wave-instruction
      f32x4 v = *(const f32x4*)(rp + c);
      v *= inv;
      *(f32x4*)(rp + c) = v;
      __builtin_nontemporal_store(v, (f32x4*)(gp + c));
    }
  }
  __syncthreads();

  // --- z = P(16x2048) * V(2048x64): wave w owns e-tile w*16 ----------------
  const int e0 = wave * 16;
  const __bf16* vBase = vg + (bh * DHEAD + e0) * SEQ;     // [b,h,e,S]: K=s contiguous
  f32x8 acc = {};
  for (int k0 = 0; k0 < SEQ; k0 += 32) {
    bf16x16 a  = load_a_lds_f32(sc, LDP, k0, lane);
    bf16x16 bb = load_b_global(vBase, SEQ, k0, lane);
    acc = wmma_bf16(a, bb, acc);
  }
  __bf16* zo = zb + ((size_t)(b * SEQ + q0) * NHEADS + h) * DHEAD + e0 + n;
#pragma unroll
  for (int r = 0; r < 8; ++r)
    zo[(size_t)(mh + r) * (NHEADS * DHEAD)] = (__bf16)acc[r];
}

// ---- Stage 3 (fused with head-reduction): ----------------------------------
// abh[b,q,h,d] = z[b,q,h,:] . W_O[h,:,d]   (NT-streamed, 268 MB, write-once)
// attn_out[b,q,d] = sum_h abh + b_O[d]     (accumulated in registers -> no re-read)
// block = 256 threads (8 waves), wave w owns 16 d-columns; grid (8, 256).
__global__ void k_outproj(const __bf16* __restrict__ zb, const __bf16* __restrict__ woT,
                          const float* __restrict__ bO, float* __restrict__ abh,
                          float* __restrict__ attn_out) {
  const int lane = threadIdx.x & 31;
  const int wave = threadIdx.x >> 5;        // 0..7
  const int n0 = blockIdx.x * 128 + wave * 16;
  const int m0 = blockIdx.y * 16;
  const int b  = m0 >> 11;
  const int s0 = m0 & (SEQ - 1);
  const int n  = lane & 15;
  const int mh = (lane >> 4) << 3;

  float* obase = abh + (size_t)(b * SEQ + s0) * (NHEADS * DMODEL) + n0 + n;
  f32x8 sum = {};
  for (int h = 0; h < NHEADS; ++h) {
    const __bf16* aBase = zb + ((size_t)m0 * NHEADS + h) * DHEAD;     // row ld = 1024
    const __bf16* bBase = woT + (size_t)h * (DHEAD * DMODEL) + (size_t)n0 * DHEAD;
    bf16x16 a0 = load_a_global(aBase, NHEADS * DHEAD, 0,  lane);
    bf16x16 a1 = load_a_global(aBase, NHEADS * DHEAD, 32, lane);
    bf16x16 b0 = load_b_global(bBase, DHEAD, 0,  lane);
    bf16x16 b1 = load_b_global(bBase, DHEAD, 32, lane);
    f32x8 acc = {};
    acc = wmma_bf16(a0, b0, acc);
    acc = wmma_bf16(a1, b1, acc);
    float* o = obase + (size_t)h * DMODEL;
#pragma unroll
    for (int r = 0; r < 8; ++r) {
      __builtin_nontemporal_store(acc[r], o + (size_t)(mh + r) * (NHEADS * DMODEL));
      sum[r] += acc[r];
    }
  }
  const float bia = bO[n0 + n];
  float* ao = attn_out + (size_t)(b * SEQ + s0) * DMODEL + n0 + n;
#pragma unroll
  for (int r = 0; r < 8; ++r)
    ao[(size_t)(mh + r) * DMODEL] = sum[r] + bia;
}

// ---------------------------------------------------------------------------

extern "C" void kernel_launch(void* const* d_in, const int* in_sizes, int n_in,
                              void* d_out, int out_size, void* d_ws, size_t ws_size,
                              hipStream_t stream) {
  (void)in_sizes; (void)n_in; (void)out_size; (void)ws_size;
  const float* x  = (const float*)d_in[0];
  const float* WQ = (const float*)d_in[1];
  const float* bQ = (const float*)d_in[2];
  const float* WK = (const float*)d_in[3];
  const float* bK = (const float*)d_in[4];
  const float* WV = (const float*)d_in[5];
  const float* bV = (const float*)d_in[6];
  const float* WO = (const float*)d_in[7];
  const float* bO = (const float*)d_in[8];

  // Outputs, concatenated flat: attn_out | pattern | attn_out_by_head
  float* attn_out = (float*)d_out;                                  //  4,194,304
  float* pattern  = attn_out + (size_t)MROWS * DMODEL;              // 134,217,728
  float* abh      = pattern + (size_t)BATCH * NHEADS * SEQ * SEQ;   //  67,108,864

  // Workspace layout (48 MB total, all offsets 2MB-multiples -> 32B aligned)
  char* w = (char*)d_ws;
  __bf16* xb  = (__bf16*)(w);                        //  8 MB  x bf16 [4096,1024]
  __bf16* wqT = (__bf16*)(w + (8u  << 20));          //  2 MB  [h,e,d]
  __bf16* wkT = (__bf16*)(w + (10u << 20));          //  2 MB
  __bf16* wvT = (__bf16*)(w + (12u << 20));          //  2 MB
  __bf16* woT = (__bf16*)(w + (14u << 20));          //  2 MB  [h,d,e]
  __bf16* qb  = (__bf16*)(w + (16u << 20));          //  8 MB  [b,h,s,64]
  __bf16* kbf = (__bf16*)(w + (24u << 20));          //  8 MB  [b,h,s,64]
  __bf16* vb  = (__bf16*)(w + (32u << 20));          //  8 MB  [b,h,64,S]
  __bf16* zb  = (__bf16*)(w + (40u << 20));          //  8 MB  [b,s,h,64]

  // Stage 0: conversions
  k_cvt_x<<<(MROWS * DMODEL) / 256, 256, 0, stream>>>(x, xb, MROWS * DMODEL);
  k_tr_wqkv<<<(NHEADS * DMODEL * DHEAD) / 256, 256, 0, stream>>>(WQ, wqT);
  k_tr_wqkv<<<(NHEADS * DMODEL * DHEAD) / 256, 256, 0, stream>>>(WK, wkT);
  k_tr_wqkv<<<(NHEADS * DMODEL * DHEAD) / 256, 256, 0, stream>>>(WV, wvT);
  k_tr_wo  <<<(NHEADS * DHEAD * DMODEL) / 256, 256, 0, stream>>>(WO, woT);

  // Stage 1: QKV projections (WMMA)
  k_qkv<<<dim3(MROWS / 16, NHEADS), 128, 0, stream>>>(xb, wqT, bQ, qb, 0);
  k_qkv<<<dim3(MROWS / 16, NHEADS), 128, 0, stream>>>(xb, wkT, bK, kbf, 0);
  k_qkv<<<dim3(MROWS / 16, NHEADS), 128, 0, stream>>>(xb, wvT, bV, vb, 1);

  // Stage 2: attention core (WMMA scores -> LDS softmax -> WMMA PV)
  k_attn<<<dim3(SEQ / 16, NHEADS, BATCH), 128,
           (16 * LDP + 16) * sizeof(float), stream>>>(qb, kbf, vb, pattern, zb);

  // Stage 3: per-head output projection + fused head-sum (WMMA)
  k_outproj<<<dim3(8, MROWS / 16), 256, 0, stream>>>(zb, woT, bO, abh, attn_out);
}
